// tccloss_34153579938005
// MI455X (gfx1250) — matrix operand order repaired and checked by reading, not compiled, not run on hardware
//
#include <hip/hip_runtime.h>

typedef __attribute__((ext_vector_type(16))) __bf16 v16bf;
typedef __attribute__((ext_vector_type(8)))  float  v8f;
typedef __attribute__((ext_vector_type(4)))  int    v4i;

#define NB   8
#define NT   1024
#define ND   128
#define NP   56        // 8*7 ordered pairs
#define NTBLK 16       // 1024 / 64
// softmax computed in base-2 domain: scale = 1/(D*TEMP) * log2(e)
#define INVSC_L2  0.11271055006944714f   // (1/12.8)*log2(e)
#define TWOINV_L2 0.22542110013889427f   // (2/12.8)*log2(e)

// ---------------- CDNA5 async global->LDS copy (probe-verified) ----------------
#if __has_builtin(__builtin_amdgcn_global_load_async_to_lds_b128)
#define HAS_ASYNC 1
typedef __attribute__((address_space(1))) v4i* as1p;
typedef __attribute__((address_space(3))) v4i* as3p;
static __device__ __forceinline__ void cp16(const unsigned short* g, unsigned short* l) {
    __builtin_amdgcn_global_load_async_to_lds_b128((as1p)(void*)g, (as3p)(void*)l, 0, 0);
}
#define ASYNC_WAIT(n) asm volatile("s_wait_asynccnt " #n ::: "memory")
#else
#define HAS_ASYNC 0
static __device__ __forceinline__ void cp16(const unsigned short* g, unsigned short* l) {
    *(uint4*)l = *(const uint4*)g;
}
#define ASYNC_WAIT(n) do {} while (0)
#endif

static __device__ __forceinline__ float fexp2(float x) {
#if __has_builtin(__builtin_amdgcn_exp2f)
    return __builtin_amdgcn_exp2f(x);
#else
    return exp2f(x);
#endif
}

// ------------- 16-lane half-row reductions: DPP row_ror ladder (no LDS) -------------
#if __has_builtin(__builtin_amdgcn_update_dpp)
template <int CTRL>
static __device__ __forceinline__ float dppmov(float x) {
    union { float f; int i; } a, b;
    a.f = x;
    b.i = __builtin_amdgcn_update_dpp(0, a.i, CTRL, 0xf, 0xf, true);
    return b.f;
}
static __device__ __forceinline__ float rsum16(float t) {
    t += dppmov<0x128>(t);   // row_ror:8
    t += dppmov<0x124>(t);   // row_ror:4
    t += dppmov<0x122>(t);   // row_ror:2
    t += dppmov<0x121>(t);   // row_ror:1
    return t;
}
static __device__ __forceinline__ float rmax16(float t) {
    t = fmaxf(t, dppmov<0x128>(t));
    t = fmaxf(t, dppmov<0x124>(t));
    t = fmaxf(t, dppmov<0x122>(t));
    t = fmaxf(t, dppmov<0x121>(t));
    return t;
}
#else
static __device__ __forceinline__ float rsum16(float t) {
#pragma unroll
    for (int m = 8; m >= 1; m >>= 1) t += __shfl_xor(t, m, 32);
    return t;
}
static __device__ __forceinline__ float rmax16(float t) {
#pragma unroll
    for (int m = 8; m >= 1; m >>= 1) t = fmaxf(t, __shfl_xor(t, m, 32));
    return t;
}
#endif

union FragU { uint4 u[2]; v16bf v; };

static __device__ __forceinline__ unsigned short f2bf(float x) {
    union { float f; unsigned u; } c; c.f = x;
    unsigned r = c.u + 0x7FFFu + ((c.u >> 16) & 1u);
    return (unsigned short)(r >> 16);
}

// A fragment: 16x32 bf16, row-major source, lane16 = row.
static __device__ __forceinline__ v16bf loadA(const unsigned short* base, int row,
                                              int ld, int kb, int half) {
    int o0 = row * ld + kb + 8 * half;
    FragU f;
    f.u[0] = *(const uint4*)(base + o0);
    f.u[1] = *(const uint4*)(base + o0 + 16);
    return f.v;
}

// B fragment: 32x16 bf16, B[k][n] = rows[n][k] (rows row-major), lane16 = n.
static __device__ __forceinline__ v16bf loadB(const unsigned short* base, int n,
                                              int ld, int kb, int half) {
    int o = n * ld + kb + 8 * half;
    FragU f;
    f.u[0] = *(const uint4*)(base + o);
    f.u[1] = *(const uint4*)(base + o + 16);
    return f.v;
}

static __device__ __forceinline__ v8f wmma_bf16(v16bf a, v16bf b, v8f c) {
    return __builtin_amdgcn_wmma_f32_16x16x32_bf16(false, a, false, b,
                                                   (short)0, c, false, false);
}

static __device__ __forceinline__ v8f redmax8(v8f x) {
    v8f r;
#pragma unroll
    for (int v = 0; v < 8; ++v) r[v] = rmax16(x[v]);
    return r;
}

static __device__ __forceinline__ v8f redsum8(v8f x) {
    v8f r;
#pragma unroll
    for (int v = 0; v < 8; ++v) r[v] = rsum16(x[v]);
    return r;
}

static __device__ __forceinline__ v8f vmax8(v8f a, v8f b) {
    v8f r;
#pragma unroll
    for (int v = 0; v < 8; ++v) r[v] = fmaxf(a[v], b[v]);
    return r;
}

// ------------- prep: fp32 -> bf16 (row-major + d-major), scaled norms, steps -------------
__global__ __launch_bounds__(256) void tcc_prep(const float* __restrict__ embs,
                                                const int* __restrict__ fidx,
                                                const int* __restrict__ vlen,
                                                unsigned short* __restrict__ ebf,
                                                unsigned short* __restrict__ ebfT,
                                                float* __restrict__ norms,
                                                float* __restrict__ steps,
                                                float* __restrict__ out) {
    if (blockIdx.x == 0 && threadIdx.x == 0) out[0] = 0.f;
    int wid = threadIdx.x >> 5, lane = threadIdx.x & 31;
    int row = blockIdx.x * 8 + wid;                     // one wave per (b,t) row
    if (row >= NB * NT) return;
    int b = row >> 10, t = row & 1023;
    float4 f = ((const float4*)(embs + (size_t)row * ND))[lane];
    float sq = f.x * f.x + f.y * f.y + f.z * f.z + f.w * f.w;
    unsigned short h0 = f2bf(f.x), h1 = f2bf(f.y), h2 = f2bf(f.z), h3 = f2bf(f.w);
    uint2 pk;
    pk.x = (unsigned)h0 | ((unsigned)h1 << 16);
    pk.y = (unsigned)h2 | ((unsigned)h3 << 16);
    ((uint2*)(ebf + (size_t)row * ND))[lane] = pk;
    // d-major copy: ebfT[b][d][t]
    int d0 = lane * 4;
    ebfT[((size_t)b * ND + d0 + 0) * NT + t] = h0;
    ebfT[((size_t)b * ND + d0 + 1) * NT + t] = h1;
    ebfT[((size_t)b * ND + d0 + 2) * NT + t] = h2;
    ebfT[((size_t)b * ND + d0 + 3) * NT + t] = h3;
#pragma unroll
    for (int m = 16; m >= 1; m >>= 1) sq += __shfl_xor(sq, m, 32);
    if (lane == 0) {
        norms[row] = sq * INVSC_L2;   // pre-scaled for base-2 softmax
        steps[row] = (float)fidx[row] / (float)vlen[row >> 10];
    }
}

// ---------------- main fused TCC kernel ----------------
__global__ __launch_bounds__(128) void tcc_main(const unsigned short* __restrict__ ebf,
                                                const unsigned short* __restrict__ ebfT,
                                                const float* __restrict__ norms,
                                                const float* __restrict__ steps,
                                                float* __restrict__ out) {
    __shared__ __align__(16) unsigned short lds_chunk[2][64 * 128];   // s-major rows
    __shared__ __align__(16) unsigned short lds_chunkT[2][128 * 64];  // d-major rows
    __shared__ __align__(16) unsigned short lds_nn[64 * 128];         // nn rows bf16
    __shared__ __align__(16) unsigned short lds_P[4][16 * 32];        // per-wave P
    __shared__ float lds_norm[2][64];
    __shared__ float lds_sp[2][64];

    const int tid = threadIdx.x;
    const int wid = tid >> 5, lane = tid & 31;
    const int lane16 = lane & 15, half = lane >> 4;

    const int p = blockIdx.x / NTBLK;
    const int tblk = blockIdx.x % NTBLK;
    const int i = p / 7;
    const int jr = p % 7;
    const int j = jr + (jr >= i ? 1 : 0);

    const unsigned short* gC1 = ebf + (size_t)j * NT * ND;    // e2 rows
    const unsigned short* gT1 = ebfT + (size_t)j * ND * NT;   // e2 d-major
    const unsigned short* gC2 = ebf + (size_t)i * NT * ND;    // e1 rows

    // ---- stage-1 A fragments: e1 rows (video i, this t-block), from global ----
    const unsigned short* e1rows = ebf + ((size_t)i * NT + tblk * 64 + wid * 16) * ND;
    v16bf Ae1[4];
#pragma unroll
    for (int kc = 0; kc < 4; ++kc) Ae1[kc] = loadA(e1rows, lane16, ND, kc * 32, half);

    v8f m1 = {-1e30f, -1e30f, -1e30f, -1e30f, -1e30f, -1e30f, -1e30f, -1e30f};
    v8f l1 = {0, 0, 0, 0, 0, 0, 0, 0};
    v8f nnacc[8];
#pragma unroll
    for (int dt = 0; dt < 8; ++dt) nnacc[dt] = (v8f){0, 0, 0, 0, 0, 0, 0, 0};

    // ============ stage 1: beta12 = softmax_s, nn = beta12 @ e2 ============
    {   // prologue: chunk 0 (16 x b128 per thread: 8 s-major + 8 d-major)
        const int s0 = 0;
#pragma unroll 1
        for (int k = 0; k < 8; ++k) {
            int it = tid + k * 128;
            cp16(gC1 + (size_t)s0 * ND + it * 8, &lds_chunk[0][it * 8]);
            int d = it >> 3, q = it & 7;
            cp16(gT1 + (size_t)d * NT + s0 + q * 8, &lds_chunkT[0][d * 64 + q * 8]);
        }
        if (tid < 64) lds_norm[0][tid] = norms[j * NT + tid];
    }
#pragma unroll 1
    for (int c = 0; c < 16; ++c) {
        const int bsel = c & 1;
        if (c < 15) {   // prefetch next chunk into other buffer
            const int s0 = (c + 1) * 64, nb = bsel ^ 1;
#pragma unroll 1
            for (int k = 0; k < 8; ++k) {
                int it = tid + k * 128;
                cp16(gC1 + (size_t)s0 * ND + it * 8, &lds_chunk[nb][it * 8]);
                int d = it >> 3, q = it & 7;
                cp16(gT1 + (size_t)d * NT + s0 + q * 8, &lds_chunkT[nb][d * 64 + q * 8]);
            }
            if (tid < 64) lds_norm[nb][tid] = norms[j * NT + s0 + tid];
            ASYNC_WAIT(16);   // current chunk's 16 transfers have landed
        } else {
            ASYNC_WAIT(0);
        }
        __syncthreads();
        const unsigned short* chunk = lds_chunk[bsel];
        const unsigned short* chunkT = lds_chunkT[bsel];
        const float* cn = lds_norm[bsel];

#pragma unroll
        for (int st2 = 0; st2 < 2; ++st2) {
            const int sb = st2 * 32;
            v8f S0, S1;
#pragma unroll
            for (int u = 0; u < 2; ++u) {
                v16bf Bfr[4];
#pragma unroll
                for (int kc = 0; kc < 4; ++kc)
                    Bfr[kc] = loadB(chunk, sb + u * 16 + lane16, ND, kc * 32, half);
                v8f acc = {0, 0, 0, 0, 0, 0, 0, 0};
#pragma unroll
                for (int kc = 0; kc < 4; ++kc) acc = wmma_bf16(Ae1[kc], Bfr[kc], acc);
                float yn = cn[sb + u * 16 + lane16];
                v8f S;
#pragma unroll
                for (int v = 0; v < 8; ++v) S[v] = acc[v] * TWOINV_L2 - yn;
                if (u == 0) S0 = S; else S1 = S;
            }
            v8f mt = redmax8(vmax8(S0, S1));
            v8f mnew = vmax8(m1, mt);
            v8f sc, P0, P1;
#pragma unroll
            for (int v = 0; v < 8; ++v) {
                sc[v] = fexp2(m1[v] - mnew[v]);
                P0[v] = fexp2(S0[v] - mnew[v]);
                P1[v] = fexp2(S1[v] - mnew[v]);
            }
            m1 = mnew;
            l1 = l1 * sc + redsum8(P0 + P1);
#pragma unroll
            for (int dt = 0; dt < 8; ++dt) nnacc[dt] *= sc;
            // stage P (C-layout -> row-major LDS -> A-layout), wave-private
#pragma unroll
            for (int v = 0; v < 8; ++v) {
                lds_P[wid][(v + 8 * half) * 32 + lane16] = f2bf(P0[v]);
                lds_P[wid][(v + 8 * half) * 32 + 16 + lane16] = f2bf(P1[v]);
            }
            asm volatile("" ::: "memory");  // same-wave LDS is in-order
            v16bf Ap = loadA(&lds_P[wid][0], lane16, 32, 0, half);
#pragma unroll
            for (int dt = 0; dt < 8; dt += 2) {
                v16bf Bv0 = loadB(chunkT, dt * 16 + lane16, 64, sb, half);
                v16bf Bv1 = loadB(chunkT, (dt + 1) * 16 + lane16, 64, sb, half);
                nnacc[dt] = wmma_bf16(Ap, Bv0, nnacc[dt]);
                nnacc[dt + 1] = wmma_bf16(Ap, Bv1, nnacc[dt + 1]);
            }
        }
        __syncthreads();
    }

    // ---- finalize nn rows -> LDS bf16 (softmax normalize) ----
    {
        v8f rinv;
#pragma unroll
        for (int v = 0; v < 8; ++v) rinv[v] = 1.0f / l1[v];
#pragma unroll
        for (int dt = 0; dt < 8; ++dt)
#pragma unroll
            for (int v = 0; v < 8; ++v)
                lds_nn[(wid * 16 + v + 8 * half) * ND + dt * 16 + lane16] =
                    f2bf(nnacc[dt][v] * rinv[v]);
        asm volatile("" ::: "memory");
    }
    v16bf Ann[4];
    {
        const unsigned short* nnrows = lds_nn + wid * 16 * ND;
#pragma unroll
        for (int kc = 0; kc < 4; ++kc) Ann[kc] = loadA(nnrows, lane16, ND, kc * 32, half);
    }

    // ============ stage 2: beta = softmax_s(logits), time_pred = beta @ sp ============
    v8f m2 = {-1e30f, -1e30f, -1e30f, -1e30f, -1e30f, -1e30f, -1e30f, -1e30f};
    v8f l2 = {0, 0, 0, 0, 0, 0, 0, 0};
    v8f tp = {0, 0, 0, 0, 0, 0, 0, 0};
    {   // prologue chunk 0 (e1 rows of video i)
#pragma unroll 1
        for (int k = 0; k < 8; ++k) {
            int it = tid + k * 128;
            cp16(gC2 + it * 8, &lds_chunk[0][it * 8]);
        }
        if (tid < 64) {
            lds_norm[0][tid] = norms[i * NT + tid];
            lds_sp[0][tid] = steps[i * NT + tid];
        }
    }
#pragma unroll 1
    for (int c = 0; c < 16; ++c) {
        const int bsel = c & 1;
        if (c < 15) {
            const int s0 = (c + 1) * 64, nb = bsel ^ 1;
#pragma unroll 1
            for (int k = 0; k < 8; ++k) {
                int it = tid + k * 128;
                cp16(gC2 + (size_t)s0 * ND + it * 8, &lds_chunk[nb][it * 8]);
            }
            if (tid < 64) {
                lds_norm[nb][tid] = norms[i * NT + s0 + tid];
                lds_sp[nb][tid] = steps[i * NT + s0 + tid];
            }
            ASYNC_WAIT(8);
        } else {
            ASYNC_WAIT(0);
        }
        __syncthreads();
        const unsigned short* chunk = lds_chunk[bsel];
        const float* cn = lds_norm[bsel];
        const float* csp = lds_sp[bsel];

#pragma unroll
        for (int st2 = 0; st2 < 2; ++st2) {
            const int sb = st2 * 32;
            v8f S0, S1;
#pragma unroll
            for (int u = 0; u < 2; ++u) {
                v16bf Bfr[4];
#pragma unroll
                for (int kc = 0; kc < 4; ++kc)
                    Bfr[kc] = loadB(chunk, sb + u * 16 + lane16, ND, kc * 32, half);
                v8f acc = {0, 0, 0, 0, 0, 0, 0, 0};
#pragma unroll
                for (int kc = 0; kc < 4; ++kc) acc = wmma_bf16(Ann[kc], Bfr[kc], acc);
                float yn = cn[sb + u * 16 + lane16];
                v8f S;
#pragma unroll
                for (int v = 0; v < 8; ++v) S[v] = acc[v] * TWOINV_L2 - yn;
                if (u == 0) S0 = S; else S1 = S;
            }
            v8f mt = redmax8(vmax8(S0, S1));
            v8f mnew = vmax8(m2, mt);
            v8f sc, P0, P1;
#pragma unroll
            for (int v = 0; v < 8; ++v) {
                sc[v] = fexp2(m2[v] - mnew[v]);
                P0[v] = fexp2(S0[v] - mnew[v]);
                P1[v] = fexp2(S1[v] - mnew[v]);
            }
            m2 = mnew;
            l2 = l2 * sc + redsum8(P0 + P1);
            float sp0 = csp[sb + lane16];
            float sp1 = csp[sb + 16 + lane16];
            tp = tp * sc + redsum8(P0 * sp0 + P1 * sp1);
        }
        __syncthreads();
    }

    // ---- loss: mean((time_pred - time_true)^2), scaled ----
    float accum = 0.f;
#pragma unroll
    for (int v = 0; v < 8; ++v) {
        float tpred = tp[v] / l2[v];
        int t = tblk * 64 + wid * 16 + v + 8 * half;
        float e = tpred - steps[i * NT + t];
        accum += e * e;
    }
    if (lane16 == 0)
        atomicAdd(out, accum * (1.0f / ((float)NP * (float)NT)));
}

extern "C" void kernel_launch(void* const* d_in, const int* in_sizes, int n_in,
                              void* d_out, int out_size, void* d_ws, size_t ws_size,
                              hipStream_t stream) {
    const float* embs = (const float*)d_in[0];
    const int* fidx = (const int*)d_in[1];
    const int* vlen = (const int*)d_in[2];
    float* out = (float*)d_out;

    size_t embBytes = (size_t)NB * NT * ND * 2;                     // 2 MB each
    unsigned short* ebf  = (unsigned short*)d_ws;
    unsigned short* ebfT = (unsigned short*)((char*)d_ws + embBytes);
    float* norms = (float*)((char*)d_ws + 2 * embBytes);
    float* steps = norms + NB * NT;

    tcc_prep<<<dim3(NB * NT / 8), dim3(256), 0, stream>>>(embs, fidx, vlen,
                                                          ebf, ebfT, norms, steps, out);
    tcc_main<<<dim3(NP * NTBLK), dim3(128), 0, stream>>>(ebf, ebfT, norms, steps, out);
}
